// Self_AttentionFlow_Layer_17360257810686
// MI455X (gfx1250) — matrix operand order repaired and checked by reading, not compiled, not run on hardware
//
#include <hip/hip_runtime.h>

// Problem constants (reference: B=16, L=1024, D=1024)
constexpr int BB = 16;
constexpr int LL = 1024;
constexpr int DD = 1024;
constexpr float NEGV = -1e10f;

// ---------------- bf16 helpers (round-to-nearest-even) ----------------
__device__ inline unsigned short f2bf(float f) {
  unsigned int u = __builtin_bit_cast(unsigned int, f);
  unsigned int r = u + 0x7FFFu + ((u >> 16) & 1u);
  return (unsigned short)(r >> 16);
}
__device__ inline float bf2f(unsigned short h) {
  unsigned int u = ((unsigned int)h) << 16;
  return __builtin_bit_cast(float, u);
}

// ---------------- WMMA fragment types (gfx1250, wave32) ----------------
typedef __attribute__((ext_vector_type(16))) __bf16 v16bf;
typedef __attribute__((ext_vector_type(8)))  float  v8f;

union ABFrag { v16bf v; unsigned short s[16]; };
union CFrag  { v8f   v; float f[8]; };

// LDS tile strides (ushort elements), padded for conflict-free fragment reads
// (row byte strides 80 / 144 are multiples of 16 -> b128 async LDS writes legal)
constexpr int LDA = 40;  // [rows][32] tiles (A side / transposed-B side)
constexpr int LDB = 72;  // [32][64] tiles (B side, K-major)

// -------- async global -> LDS staging (gfx1250 async path, ASYNCcnt) -------
// Builtin signature (from hipcc diagnostic): arg0 is a pointer to a 4xi32
// vector in the global address space; arg1 the LDS-side pointer; then
// (imm offset, cpol).
typedef __attribute__((__vector_size__(4 * sizeof(int)))) int v4i_t;
typedef v4i_t __attribute__((address_space(1)))* gas_v4i_ptr;
typedef v4i_t __attribute__((address_space(3)))* las_v4i_ptr;

__device__ inline void cp_async_b128(const unsigned short* g, unsigned short* l) {
  __builtin_amdgcn_global_load_async_to_lds_b128(
      (gas_v4i_ptr)g, (las_v4i_ptr)l, 0, 0);
}
__device__ inline void wait_async_all() { __builtin_amdgcn_s_wait_asynccnt(0); }

// A tile: 128 rows x 32 cols; 256 threads -> two b128 copies per thread.
__device__ inline void asyncA(const unsigned short* g, int rowStride,
                              unsigned short* lds, int tid) {
  int r = tid >> 1, q = tid & 1;
  const unsigned short* gp = g + (size_t)r * rowStride + q * 16;
  unsigned short* lp = lds + r * LDA + q * 16;     // byte 80r + 32q : 16B aligned
  cp_async_b128(gp, lp);
  cp_async_b128(gp + 8, lp + 8);
}
// B tile, transposed layout: 64 rows x 32 cols (stride LDA); one b128/thread.
__device__ inline void asyncB64(const unsigned short* g, int rowStride,
                                unsigned short* lds, int tid) {
  int r = tid >> 2, q = tid & 3;
  cp_async_b128(g + (size_t)r * rowStride + q * 8,
                lds + r * LDA + q * 8);            // byte 80r + 16q : 16B aligned
}
// B tile, K-major: 32 rows x 64 cols (stride LDB); one b128/thread.
__device__ inline void asyncB32(const unsigned short* g, int rowStride,
                                unsigned short* lds, int tid) {
  int r = tid >> 3, q = tid & 7;
  cp_async_b128(g + (size_t)r * rowStride + q * 8,
                lds + r * LDB + q * 8);            // byte 144r + 16q : 16B aligned
}

// A-matrix 16x32 bf16 fragment from LDS tile[row][k], row stride LDA.
// Lanes 0-15: rows M=0..15, K 0..15; lanes 16-31: K 16..31.
__device__ inline void load_a_frag(const unsigned short* tile, int row0, int lane, ABFrag& a) {
  int r  = row0 + (lane & 15);
  int hk = (lane >> 4) ? 16 : 0;
  const unsigned short* p = tile + r * LDA;
#pragma unroll
  for (int v = 0; v < 8; ++v) {
    int k = 2 * v + hk;
    a.s[2 * v]     = p[k];
    a.s[2 * v + 1] = p[k + 1];
  }
}

// B-matrix 32x16 bf16 fragment from LDS tile stored [k][n], row stride LDB.
__device__ inline void load_b_frag_kn(const unsigned short* tile, int n0, int lane, ABFrag& b) {
  int n  = n0 + (lane & 15);
  int kh = (lane >> 4) ? 16 : 0;
#pragma unroll
  for (int v = 0; v < 8; ++v) {
    int k = kh + 2 * v;
    b.s[2 * v]     = tile[k * LDB + n];
    b.s[2 * v + 1] = tile[(k + 1) * LDB + n];
  }
}

// Same B fragment from a transposed tile [n][k] (stride LDA) — used when both
// GEMM operands are row-major in K (scores = Y @ X^T).
__device__ inline void load_b_frag_nk(const unsigned short* tile, int n0, int lane, ABFrag& b) {
  int n  = n0 + (lane & 15);
  int kh = (lane >> 4) ? 16 : 0;
  const unsigned short* p = tile + n * LDA;
#pragma unroll
  for (int v = 0; v < 8; ++v) {
    int k = kh + 2 * v;
    b.s[2 * v]     = p[k];
    b.s[2 * v + 1] = p[k + 1];
  }
}

// =====================================================================
// Kernel 1: bf16 casts + per-row dot products s_c, s_q
// =====================================================================
__global__ __launch_bounds__(256) void prep_kernel(
    const float* __restrict__ inp, const float* __restrict__ w_sim,
    unsigned short* __restrict__ X, unsigned short* __restrict__ Y,
    float* __restrict__ sc, float* __restrict__ sq) {
  int row = blockIdx.x;
  const float* src = inp + (size_t)row * DD;
  float c = 0.f, q = 0.f;
#pragma unroll
  for (int t = 0; t < 4; ++t) {
    int d = threadIdx.x + t * 256;
    float x = src[d];
    X[(size_t)row * DD + d] = f2bf(x);
    Y[(size_t)row * DD + d] = f2bf(x * w_sim[2 * DD + d]);
    c += x * w_sim[d];
    q += x * w_sim[DD + d];
  }
#pragma unroll
  for (int off = 16; off; off >>= 1) {
    c += __shfl_xor(c, off, 32);
    q += __shfl_xor(q, off, 32);
  }
  __shared__ float rc[8], rq[8];
  int lane = threadIdx.x & 31, w = threadIdx.x >> 5;
  if (lane == 0) { rc[w] = c; rq[w] = q; }
  __syncthreads();
  if (threadIdx.x == 0) {
    float tc = 0.f, tq = 0.f;
#pragma unroll
    for (int i = 0; i < 8; ++i) { tc += rc[i]; tq += rq[i]; }
    sc[row] = tc; sq[row] = tq;
  }
}

// =====================================================================
// Kernel 2: Wm f32 -> bf16
// =====================================================================
__global__ __launch_bounds__(256) void wconv_kernel(
    const float* __restrict__ Wm, unsigned short* __restrict__ Wb) {
  size_t i = ((size_t)blockIdx.x * 256 + threadIdx.x) * 4;
#pragma unroll
  for (int t = 0; t < 4; ++t) Wb[i + t] = f2bf(Wm[i + t]);
}

// =====================================================================
// Kernel 3: scores = Y @ X^T + s_c + s_q, masked.
// Block tile 128(i) x 64(j), 8 waves (4x2), wave tile 32x32 -> 4 WMMA/K-step.
// Double-buffered LDS fed by async global->LDS copies (ASYNCcnt).
// =====================================================================
__global__ __launch_bounds__(256) void scores_kernel(
    const unsigned short* __restrict__ X, const unsigned short* __restrict__ Y,
    const float* __restrict__ sc, const float* __restrict__ sq,
    const unsigned char* __restrict__ mask, float* __restrict__ S) {
  __shared__ __attribute__((aligned(16))) unsigned short As[2][128 * LDA];
  __shared__ __attribute__((aligned(16))) unsigned short Bs[2][64 * LDA];

  const int nbi = LL / 128, nbj = LL / 64;
  int b  = blockIdx.x / (nbi * nbj);
  int t  = blockIdx.x % (nbi * nbj);
  int i0 = (t / nbj) * 128;
  int j0 = (t % nbj) * 64;

  int tid = threadIdx.x, lane = tid & 31, w = tid >> 5;
  int wm = (w >> 1) * 32, wn = (w & 1) * 32;

  CFrag acc[2][2];
#pragma unroll
  for (int m = 0; m < 2; ++m)
#pragma unroll
    for (int n = 0; n < 2; ++n)
#pragma unroll
      for (int i = 0; i < 8; ++i) acc[m][n].f[i] = 0.f;

  const unsigned short* Yb = Y + (size_t)(b * LL + i0) * DD;
  const unsigned short* Xb = X + (size_t)(b * LL + j0) * DD;

  asyncA(Yb, DD, As[0], tid);
  asyncB64(Xb, DD, Bs[0], tid);

  int cur = 0;
  for (int k0 = 0; k0 < DD; k0 += 32) {
    wait_async_all();          // my copies into buf[cur] done
    __syncthreads();           // everyone's copies visible
    bool hasNext = (k0 + 32) < DD;
    int nxt = cur ^ 1;
    if (hasNext) {             // fire-and-forget copies for the next tile
      asyncA(Yb + k0 + 32, DD, As[nxt], tid);
      asyncB64(Xb + k0 + 32, DD, Bs[nxt], tid);
    }
    if (k0 + 64 < DD) {        // warm L2 two tiles ahead
      __builtin_prefetch(Yb + k0 + 64 + (size_t)(tid >> 1) * DD, 0, 3);
      __builtin_prefetch(Xb + k0 + 64 + (size_t)(tid >> 2) * DD, 0, 3);
    }
    ABFrag a0, a1, b0, b1;
    load_a_frag(As[cur], wm, lane, a0);
    load_a_frag(As[cur], wm + 16, lane, a1);
    load_b_frag_nk(Bs[cur], wn, lane, b0);
    load_b_frag_nk(Bs[cur], wn + 16, lane, b1);
    acc[0][0].v = __builtin_amdgcn_wmma_f32_16x16x32_bf16(false, a0.v, false, b0.v, (short)0, acc[0][0].v, false, false);
    acc[0][1].v = __builtin_amdgcn_wmma_f32_16x16x32_bf16(false, a0.v, false, b1.v, (short)0, acc[0][1].v, false, false);
    acc[1][0].v = __builtin_amdgcn_wmma_f32_16x16x32_bf16(false, a1.v, false, b0.v, (short)0, acc[1][0].v, false, false);
    acc[1][1].v = __builtin_amdgcn_wmma_f32_16x16x32_bf16(false, a1.v, false, b1.v, (short)0, acc[1][1].v, false, false);
    cur = nxt;
  }

  int hi  = (lane >> 4) ? 8 : 0;
  int col = lane & 15;
#pragma unroll
  for (int mi = 0; mi < 2; ++mi) {
#pragma unroll
    for (int r = 0; r < 8; ++r) {
      int i = i0 + wm + mi * 16 + r + hi;
      float base = sc[b * LL + i];
      unsigned char mrow = mask[b * LL + i];
#pragma unroll
      for (int half = 0; half < 2; ++half) {
        int j = j0 + wn + half * 16 + col;
        float v = acc[mi][half].f[r] + base + sq[b * LL + j];
        if (mrow | mask[b * LL + j]) v = NEGV;
        S[((size_t)b * LL + i) * LL + j] = v;
      }
    }
  }
}

// =====================================================================
// Kernel 4: row softmax over L, output bf16 attn
// =====================================================================
__global__ __launch_bounds__(256) void softmax_kernel(
    const float* __restrict__ S, unsigned short* __restrict__ A) {
  int row = blockIdx.x;
  const float* src = S + (size_t)row * LL;
  float v[4], mx = -3.4e38f;
#pragma unroll
  for (int t = 0; t < 4; ++t) {
    v[t] = src[threadIdx.x + t * 256];
    mx = fmaxf(mx, v[t]);
  }
#pragma unroll
  for (int off = 16; off; off >>= 1) mx = fmaxf(mx, __shfl_xor(mx, off, 32));
  __shared__ float red[8], red2[8];
  int lane = threadIdx.x & 31, w = threadIdx.x >> 5;
  if (lane == 0) red[w] = mx;
  __syncthreads();
  float bmax = red[0];
#pragma unroll
  for (int i = 1; i < 8; ++i) bmax = fmaxf(bmax, red[i]);
  float s = 0.f;
#pragma unroll
  for (int t = 0; t < 4; ++t) { v[t] = __expf(v[t] - bmax); s += v[t]; }
#pragma unroll
  for (int off = 16; off; off >>= 1) s += __shfl_xor(s, off, 32);
  if (lane == 0) red2[w] = s;
  __syncthreads();
  float tot = 0.f;
#pragma unroll
  for (int i = 0; i < 8; ++i) tot += red2[i];
  float inv = 1.f / tot;
#pragma unroll
  for (int t = 0; t < 4; ++t)
    A[(size_t)row * LL + threadIdx.x + t * 256] = f2bf(v[t] * inv);
}

// =====================================================================
// Kernel 5: c2q = attn @ X, fused epilogue writes C and XC = C*inp (bf16).
// =====================================================================
__global__ __launch_bounds__(256) void c2q_kernel(
    const unsigned short* __restrict__ attn, const unsigned short* __restrict__ X,
    unsigned short* __restrict__ C, unsigned short* __restrict__ XC) {
  __shared__ __attribute__((aligned(16))) unsigned short As[2][128 * LDA];
  __shared__ __attribute__((aligned(16))) unsigned short Bs[2][32 * LDB];

  const int nbi = LL / 128, nd = DD / 64;
  int b  = blockIdx.x / (nbi * nd);
  int t  = blockIdx.x % (nbi * nd);
  int i0 = (t / nd) * 128;
  int d0 = (t % nd) * 64;

  int tid = threadIdx.x, lane = tid & 31, w = tid >> 5;
  int wm = (w >> 1) * 32, wn = (w & 1) * 32;

  CFrag acc[2][2];
#pragma unroll
  for (int m = 0; m < 2; ++m)
#pragma unroll
    for (int n = 0; n < 2; ++n)
#pragma unroll
      for (int i = 0; i < 8; ++i) acc[m][n].f[i] = 0.f;

  const unsigned short* Ab = attn + (size_t)(b * LL + i0) * LL;
  const unsigned short* Xb = X + (size_t)b * LL * DD + d0;

  asyncA(Ab, LL, As[0], tid);
  asyncB32(Xb, DD, Bs[0], tid);

  int cur = 0;
  for (int k0 = 0; k0 < LL; k0 += 32) {
    wait_async_all();
    __syncthreads();
    bool hasNext = (k0 + 32) < LL;
    int nxt = cur ^ 1;
    if (hasNext) {
      asyncA(Ab + k0 + 32, LL, As[nxt], tid);
      asyncB32(Xb + (size_t)(k0 + 32) * DD, DD, Bs[nxt], tid);
    }
    if (k0 + 64 < LL) {
      __builtin_prefetch(Ab + k0 + 64 + (size_t)(tid >> 1) * LL, 0, 3);
      __builtin_prefetch(Xb + (size_t)(k0 + 64 + (tid >> 3)) * DD, 0, 3);
    }
    ABFrag a0, a1, b0, b1;
    load_a_frag(As[cur], wm, lane, a0);
    load_a_frag(As[cur], wm + 16, lane, a1);
    load_b_frag_kn(Bs[cur], wn, lane, b0);
    load_b_frag_kn(Bs[cur], wn + 16, lane, b1);
    acc[0][0].v = __builtin_amdgcn_wmma_f32_16x16x32_bf16(false, a0.v, false, b0.v, (short)0, acc[0][0].v, false, false);
    acc[0][1].v = __builtin_amdgcn_wmma_f32_16x16x32_bf16(false, a0.v, false, b1.v, (short)0, acc[0][1].v, false, false);
    acc[1][0].v = __builtin_amdgcn_wmma_f32_16x16x32_bf16(false, a1.v, false, b0.v, (short)0, acc[1][0].v, false, false);
    acc[1][1].v = __builtin_amdgcn_wmma_f32_16x16x32_bf16(false, a1.v, false, b1.v, (short)0, acc[1][1].v, false, false);
    cur = nxt;
  }

  int hi  = (lane >> 4) ? 8 : 0;
  int col = lane & 15;
#pragma unroll
  for (int mi = 0; mi < 2; ++mi) {
#pragma unroll
    for (int r = 0; r < 8; ++r) {
      int i = i0 + wm + mi * 16 + r + hi;
#pragma unroll
      for (int half = 0; half < 2; ++half) {
        int d = d0 + wn + half * 16 + col;
        size_t idx = ((size_t)b * LL + i) * DD + d;
        float cv = acc[mi][half].f[r];
        C[idx]  = f2bf(cv);
        XC[idx] = f2bf(cv * bf2f(X[idx]));
      }
    }
  }
}

// =====================================================================
// Kernel 6: out = relu([X|C|XC] @ Wm + bm), zero masked rows.  K = 3072.
// =====================================================================
__global__ __launch_bounds__(256) void merge_kernel(
    const unsigned short* __restrict__ Xf, const unsigned short* __restrict__ Cf,
    const unsigned short* __restrict__ XCf, const unsigned short* __restrict__ Wb,
    const float* __restrict__ bmv, const unsigned char* __restrict__ mask,
    float* __restrict__ out) {
  __shared__ __attribute__((aligned(16))) unsigned short As[2][128 * LDA];
  __shared__ __attribute__((aligned(16))) unsigned short Bs[2][32 * LDB];

  const int ncol = DD / 64;
  int m0 = (blockIdx.x / ncol) * 128;   // flat row tile over B*L
  int n0 = (blockIdx.x % ncol) * 64;

  int tid = threadIdx.x, lane = tid & 31, w = tid >> 5;
  int wm = (w >> 1) * 32, wn = (w & 1) * 32;

  CFrag acc[2][2];
#pragma unroll
  for (int m = 0; m < 2; ++m)
#pragma unroll
    for (int n = 0; n < 2; ++n)
#pragma unroll
      for (int i = 0; i < 8; ++i) acc[m][n].f[i] = 0.f;

  const unsigned short* seg_src[3] = { Xf, Cf, XCf };

  asyncA(Xf + (size_t)m0 * DD, DD, As[0], tid);
  asyncB32(Wb + n0, DD, Bs[0], tid);

  int cur = 0;
  for (int k0 = 0; k0 < 3 * DD; k0 += 32) {
    wait_async_all();
    __syncthreads();
    bool hasNext = (k0 + 32) < 3 * DD;
    int nxt = cur ^ 1;
    if (hasNext) {
      int kn = k0 + 32;
      const unsigned short* Asrc = seg_src[kn >> 10];
      asyncA(Asrc + (size_t)m0 * DD + (kn & (DD - 1)), DD, As[nxt], tid);
      asyncB32(Wb + (size_t)kn * DD + n0, DD, Bs[nxt], tid);
    }
    if (k0 + 64 < 3 * DD)
      __builtin_prefetch(Wb + (size_t)(k0 + 64 + (tid >> 3)) * DD + n0, 0, 3);
    ABFrag a0, a1, b0, b1;
    load_a_frag(As[cur], wm, lane, a0);
    load_a_frag(As[cur], wm + 16, lane, a1);
    load_b_frag_kn(Bs[cur], wn, lane, b0);
    load_b_frag_kn(Bs[cur], wn + 16, lane, b1);
    acc[0][0].v = __builtin_amdgcn_wmma_f32_16x16x32_bf16(false, a0.v, false, b0.v, (short)0, acc[0][0].v, false, false);
    acc[0][1].v = __builtin_amdgcn_wmma_f32_16x16x32_bf16(false, a0.v, false, b1.v, (short)0, acc[0][1].v, false, false);
    acc[1][0].v = __builtin_amdgcn_wmma_f32_16x16x32_bf16(false, a1.v, false, b0.v, (short)0, acc[1][0].v, false, false);
    acc[1][1].v = __builtin_amdgcn_wmma_f32_16x16x32_bf16(false, a1.v, false, b1.v, (short)0, acc[1][1].v, false, false);
    cur = nxt;
  }

  int hi  = (lane >> 4) ? 8 : 0;
  int col = lane & 15;
#pragma unroll
  for (int mi = 0; mi < 2; ++mi) {
#pragma unroll
    for (int r = 0; r < 8; ++r) {
      int row = m0 + wm + mi * 16 + r + hi;
      unsigned char m = mask[row];
#pragma unroll
      for (int half = 0; half < 2; ++half) {
        int n = n0 + wn + half * 16 + col;
        float v = acc[mi][half].f[r] + bmv[n];
        v = v > 0.f ? v : 0.f;
        if (m) v = 0.f;
        out[(size_t)row * DD + n] = v;
      }
    }
  }
}

// =====================================================================
// Workspace layout (bytes). Peak ~134 MB — scores region reused for C/XC.
// =====================================================================
constexpr size_t SZ_BF = (size_t)BB * LL * DD * 2;            // 32 MB
constexpr size_t OFF_X  = 0;
constexpr size_t OFF_Y  = OFF_X + SZ_BF;                      // reused as attn
constexpr size_t OFF_WB = OFF_Y + SZ_BF;
constexpr size_t OFF_SC = OFF_WB + (size_t)3 * DD * DD * 2;
constexpr size_t OFF_SQ = OFF_SC + (size_t)BB * LL * 4;
constexpr size_t OFF_S  = (OFF_SQ + (size_t)BB * LL * 4 + 255) & ~(size_t)255;
constexpr size_t OFF_C  = OFF_S;                              // reuse after softmax
constexpr size_t OFF_XC = OFF_S + SZ_BF;

extern "C" void kernel_launch(void* const* d_in, const int* in_sizes, int n_in,
                              void* d_out, int out_size, void* d_ws, size_t ws_size,
                              hipStream_t stream) {
  (void)in_sizes; (void)n_in; (void)out_size; (void)ws_size;
  const float*         inp   = (const float*)d_in[0];
  const unsigned char* mask  = (const unsigned char*)d_in[1];
  const float*         w_sim = (const float*)d_in[2];
  const float*         Wm    = (const float*)d_in[3];
  const float*         bmv   = (const float*)d_in[4];
  float* out = (float*)d_out;

  char* ws = (char*)d_ws;
  unsigned short* X    = (unsigned short*)(ws + OFF_X);
  unsigned short* Y    = (unsigned short*)(ws + OFF_Y);
  unsigned short* Attn = (unsigned short*)(ws + OFF_Y);   // reuses Y after scores
  unsigned short* Wb   = (unsigned short*)(ws + OFF_WB);
  float*          sc   = (float*)(ws + OFF_SC);
  float*          sq   = (float*)(ws + OFF_SQ);
  float*          S    = (float*)(ws + OFF_S);
  unsigned short* C    = (unsigned short*)(ws + OFF_C);   // reuses scores region
  unsigned short* XC   = (unsigned short*)(ws + OFF_XC);

  prep_kernel<<<BB * LL, 256, 0, stream>>>(inp, w_sim, X, Y, sc, sq);
  wconv_kernel<<<(3 * DD * DD) / 1024, 256, 0, stream>>>(Wm, Wb);
  scores_kernel<<<BB * (LL / 128) * (LL / 64), 256, 0, stream>>>(X, Y, sc, sq, mask, S);
  softmax_kernel<<<BB * LL, 256, 0, stream>>>(S, Attn);
  c2q_kernel<<<BB * (LL / 128) * (DD / 64), 256, 0, stream>>>(Attn, X, C, XC);
  merge_kernel<<<(BB * LL / 128) * (DD / 64), 256, 0, stream>>>(X, C, XC, Wb, bmv, mask, out);
}
